// Money_former_block_56693568307396
// MI455X (gfx1250) — compile-verified, matrix operand order checked by reading, not run on hardware
//
#include <hip/hip_runtime.h>
#include <hip/hip_bf16.h>
#include <math.h>

// ---------------------------------------------------------------------------
// Types for CDNA5 WMMA (wave32): bf16 A/B fragments (16 halves = 8 VGPRs),
// f32 C/D accumulator (8 VGPRs).
// ---------------------------------------------------------------------------
typedef __bf16 bf16_t;
typedef __attribute__((ext_vector_type(16))) __bf16 v16bf;
typedef __attribute__((ext_vector_type(8)))  __bf16 v8bf;
typedef __attribute__((ext_vector_type(8)))  float  v8f;
typedef __attribute__((ext_vector_type(4)))  int    v4i;

static constexpr int D_MODEL = 1024;
static constexpr int NHEAD   = 16;
static constexpr int HEAD_DIM= 64;
static constexpr int D_FF    = 4096;
static constexpr int Bsz     = 2;
static constexpr int S       = 2048;
static constexpr int Mrows   = Bsz * S;          // 4096

// gfx1250 async global->LDS copy (ASYNCcnt-tracked), if the toolchain has it.
// Probe-observed parameter 0 type: AS1 pointer to int ext_vector(4).
#if defined(__HIP_DEVICE_COMPILE__) && __has_builtin(__builtin_amdgcn_global_load_async_to_lds_b128)
#define HAVE_ASYNC_LDS 1
typedef __attribute__((address_space(1))) v4i* glb_v4i_ptr;
typedef __attribute__((address_space(3))) v4i* lds_v4i_ptr;
#else
#define HAVE_ASYNC_LDS 0
#endif

__device__ inline void wait_asynccnt0() {
#if __has_builtin(__builtin_amdgcn_s_wait_asynccnt)
  __builtin_amdgcn_s_wait_asynccnt(0);
#else
  asm volatile("s_wait_asynccnt 0" ::: "memory");
#endif
}

__device__ inline v8f wmma_bf16(v16bf a, v16bf b, v8f c) {
  // (neg_a, A, neg_b, B, c_mod, C, reuse_a, reuse_b)
  return __builtin_amdgcn_wmma_f32_16x16x32_bf16(false, a, false, b, (short)0, c,
                                                 false, false);
}

// Load a 16x32 bf16 A/B fragment row for this lane.
// ISA layout: lanes 0-15 hold K {0..7, 16..23}; lanes 16-31 hold K {8..15, 24..31}.
// `p` points at (row, k0); must be 16B aligned. `hi` = lane>>4.
__device__ inline v16bf load_frag(const bf16_t* p, int hi) {
  v8bf lo  = *(const v8bf*)(p + hi * 8);
  v8bf hib = *(const v8bf*)(p + 16 + hi * 8);
  v16bf r;
#pragma unroll
  for (int i = 0; i < 8; ++i) { r[i] = lo[i]; r[i + 8] = hib[i]; }
  return r;
}

__device__ inline v8f zero_v8f() {
  v8f z;
#pragma unroll
  for (int i = 0; i < 8; ++i) z[i] = 0.0f;
  return z;
}

// ---------------------------------------------------------------------------
// fp32 -> bf16 convert (weights)
// ---------------------------------------------------------------------------
__global__ void k_convert_bf16(const float* __restrict__ in, bf16_t* __restrict__ out, int n) {
  int i = blockIdx.x * blockDim.x + threadIdx.x;
  if (i < n) out[i] = (bf16_t)in[i];
}

// ---------------------------------------------------------------------------
// RMSNorm: one block per row of D_MODEL, writes bf16
// ---------------------------------------------------------------------------
__global__ void k_rmsnorm_bf16(const float* __restrict__ x, const float* __restrict__ w,
                               bf16_t* __restrict__ out, int D) {
  const int row = blockIdx.x;
  const float* xr = x + (size_t)row * D;
  float ss = 0.0f;
  for (int i = threadIdx.x; i < D; i += blockDim.x) { float v = xr[i]; ss += v * v; }
#pragma unroll
  for (int off = 16; off > 0; off >>= 1) ss += __shfl_xor(ss, off, 32);
  __shared__ float red[8];
  __shared__ float stot;
  if ((threadIdx.x & 31) == 0) red[threadIdx.x >> 5] = ss;
  __syncthreads();
  if (threadIdx.x == 0) {
    float t = 0.0f;
#pragma unroll
    for (int i = 0; i < 8; ++i) t += red[i];
    stot = t;
  }
  __syncthreads();
  float inv = rsqrtf(stot / (float)D + 1.1920929e-07f);
  for (int i = threadIdx.x; i < D; i += blockDim.x)
    out[(size_t)row * D + i] = (bf16_t)(xr[i] * inv * w[i]);
}

// ---------------------------------------------------------------------------
// WMMA GEMM: C[M,N] = A_bf16[M,K] @ W_bf16[N,K]^T + bias[N] (+ res[M,N])
// Block: 256 threads = 8 waves; wave w does rows [m0+32w, +32), cols [n0, n0+64)
// (two 16-row groups -> 8 accumulators, 8 WMMAs per k-step).
// B tile (64 x 32 halves) double-buffered in LDS, one barrier per k-step;
// next tile copy issued right after the barrier to overlap the WMMAs
// (async global->LDS where available, register staging otherwise).
// ---------------------------------------------------------------------------
__global__ void k_gemm_bf16_wmma(const bf16_t* __restrict__ A, const bf16_t* __restrict__ W,
                                 const float* __restrict__ bias, const float* __restrict__ res,
                                 float* __restrict__ C, int M, int N, int K) {
  __shared__ bf16_t Bs[2][64 * 32];
  const int lane = threadIdx.x & 31;
  const int wave = threadIdx.x >> 5;
  const int hi   = lane >> 4;
  const int ln   = lane & 15;
  const int n0 = blockIdx.x * 64;
  const int m0 = blockIdx.y * 256 + wave * 32;

  // staging slot for this thread: row n = tid>>2, 8-half chunk ch = tid&3
  const int sn  = threadIdx.x >> 2;
  const int sch = threadIdx.x & 3;
  const bf16_t* gW   = W + (size_t)(n0 + sn) * K + sch * 8;   // + kb
  const int     slot = sn * 32 + sch * 8;

  const bf16_t* Arow0 = A + (size_t)(m0 + ln) * K;
  const bf16_t* Arow1 = A + (size_t)(m0 + 16 + ln) * K;

  v8f acc0[4], acc1[4];
#pragma unroll
  for (int j = 0; j < 4; ++j) { acc0[j] = zero_v8f(); acc1[j] = zero_v8f(); }

  // preload k-tile 0 into buffer 0
#if HAVE_ASYNC_LDS
  __builtin_amdgcn_global_load_async_to_lds_b128(
      (glb_v4i_ptr)gW, (lds_v4i_ptr)&Bs[0][slot], 0, 0);
  wait_asynccnt0();
#else
  *(v8bf*)&Bs[0][slot] = *(const v8bf*)gW;
#endif

  int buf = 0;
  for (int kb = 0; kb < K; kb += 32) {
    __syncthreads();                      // prev reads of Bs[buf^1] done; Bs[buf] visible
    const bool has_next = (kb + 32) < K;
#if HAVE_ASYNC_LDS
    if (has_next)
      __builtin_amdgcn_global_load_async_to_lds_b128(
          (glb_v4i_ptr)(gW + kb + 32), (lds_v4i_ptr)&Bs[buf ^ 1][slot], 0, 0);
#else
    v8bf streg;
    if (has_next) streg = *(const v8bf*)(gW + kb + 32);
#endif

    v16bf a0 = load_frag(Arow0 + kb, hi);
    v16bf a1 = load_frag(Arow1 + kb, hi);
#pragma unroll
    for (int j = 0; j < 4; ++j) {
      v16bf bfrag = load_frag(&Bs[buf][(16 * j + ln) * 32], hi);
      acc0[j] = wmma_bf16(a0, bfrag, acc0[j]);
      acc1[j] = wmma_bf16(a1, bfrag, acc1[j]);
    }

#if HAVE_ASYNC_LDS
    if (has_next) wait_asynccnt0();       // next tile resident before next barrier
#else
    if (has_next) *(v8bf*)&Bs[buf ^ 1][slot] = streg;
#endif
    buf ^= 1;
  }

#pragma unroll
  for (int j = 0; j < 4; ++j) {
    const int n = n0 + 16 * j + ln;
    const float bn = bias ? bias[n] : 0.0f;
#pragma unroll
    for (int i = 0; i < 8; ++i) {
      const int m = m0 + i + 8 * hi;      // C layout: VGPR i -> M = i + 8*(lane/16)
      float v = acc0[j][i] + bn;
      if (res) v += res[(size_t)m * N + n];
      C[(size_t)m * N + n] = v;
      const int m2 = m + 16;
      float v2 = acc1[j][i] + bn;
      if (res) v2 += res[(size_t)m2 * N + n];
      C[(size_t)m2 * N + n] = v2;
    }
  }
}

// ---------------------------------------------------------------------------
// RoPE + fp32->bf16, layout [B,S,H,Dh]; one thread per (b,s,h,pair)
// ---------------------------------------------------------------------------
__global__ void k_rope_bf16(const float* __restrict__ x, const float* __restrict__ cosb,
                            const float* __restrict__ sinb, bf16_t* __restrict__ out,
                            int npairs) {
  int p = blockIdx.x * blockDim.x + threadIdx.x;
  if (p >= npairs) return;
  const int j = p & 31;                       // Dh/2 = 32
  const int s = (p / (32 * NHEAD)) % S;
  const float xr = x[2 * (size_t)p + 0];
  const float xi = x[2 * (size_t)p + 1];
  const float c  = cosb[s * 32 + j];
  const float sn = sinb[s * 32 + j];
  out[2 * (size_t)p + 0] = (bf16_t)(xr * c - xi * sn);
  out[2 * (size_t)p + 1] = (bf16_t)(xr * sn + xi * c);
}

// V: fp32 [B,S,H,Dh] -> bf16 transposed [B,H,Dh,S] (so P@V B-frags are contiguous)
__global__ void k_v_transpose_bf16(const float* __restrict__ v, bf16_t* __restrict__ vt, int n) {
  int i = blockIdx.x * blockDim.x + threadIdx.x;
  if (i >= n) return;
  const int d = i & 63;
  const int h = (i >> 6) & (NHEAD - 1);
  const int s = (i >> 10) & (S - 1);
  const int b = i >> 21;                      // 1024*2048 = 2^21
  vt[(((size_t)b * NHEAD + h) * HEAD_DIM + d) * S + s] = (bf16_t)v[i];
}

// ---------------------------------------------------------------------------
// Flash-style causal attention, all-WMMA. One wave per (b, h, 16-query tile).
// Key blocks of 32. Per-wave LDS holds the 16x32 P tile for the C->A transpose.
// ---------------------------------------------------------------------------
__global__ void k_attn_wmma(const bf16_t* __restrict__ qb, const bf16_t* __restrict__ kb,
                            const bf16_t* __restrict__ vt, bf16_t* __restrict__ ob) {
  __shared__ bf16_t Pl[8][16 * 32];
  const int lane = threadIdx.x & 31;
  const int wave = threadIdx.x >> 5;
  const int hi   = lane >> 4;
  const int ln   = lane & 15;

  const int nqt = S / 16;                     // 128
  const int wid = blockIdx.x * 8 + wave;
  const int qt = wid % nqt;
  const int h  = (wid / nqt) % NHEAD;
  const int b  = wid / (nqt * NHEAD);
  const int q0 = qt * 16;

  const bf16_t* qrow = qb + ((size_t)(b * S + q0 + ln) * NHEAD + h) * HEAD_DIM;
  const v16bf qf0 = load_frag(qrow + 0, hi);
  const v16bf qf1 = load_frag(qrow + 32, hi);

  float mi[8], li[8];
  v8f acc[4];
#pragma unroll
  for (int i = 0; i < 8; ++i) { mi[i] = -3.0e38f; li[i] = 0.0f; }
#pragma unroll
  for (int j = 0; j < 4; ++j) acc[j] = zero_v8f();

  const bf16_t* vbase = vt + ((size_t)(b * NHEAD + h) * HEAD_DIM) * S;
  const int kend = q0 + 16;                   // causal: keys < kend

  for (int k0 = 0; k0 < kend; k0 += 32) {
    // ---- scores S = (Q K^T) * 0.125, two 16-key C tiles
    const bf16_t* kr0 = kb + ((size_t)(b * S + k0 + ln) * NHEAD + h) * HEAD_DIM;
    const bf16_t* kr1 = kb + ((size_t)(b * S + k0 + 16 + ln) * NHEAD + h) * HEAD_DIM;
    v8f s0 = zero_v8f(), s1 = zero_v8f();
    s0 = wmma_bf16(qf0, load_frag(kr0, hi),      s0);
    s0 = wmma_bf16(qf1, load_frag(kr0 + 32, hi), s0);
    s1 = wmma_bf16(qf0, load_frag(kr1, hi),      s1);
    s1 = wmma_bf16(qf1, load_frag(kr1 + 32, hi), s1);

    // ---- online softmax (row = i + 8*hi; row values live across 16-lane half)
#pragma unroll
    for (int i = 0; i < 8; ++i) {
      const int qidx = q0 + i + 8 * hi;
      float a0 = (k0 + ln      <= qidx) ? s0[i] * 0.125f : -1.0e9f;
      float a1 = (k0 + 16 + ln <= qidx) ? s1[i] * 0.125f : -1.0e9f;
      float mx = fmaxf(a0, a1);
#pragma unroll
      for (int msk = 1; msk <= 8; msk <<= 1) mx = fmaxf(mx, __shfl_xor(mx, msk, 32));
      const float mnew = fmaxf(mi[i], mx);
      const float e0 = __expf(a0 - mnew);
      const float e1 = __expf(a1 - mnew);
      float rs = e0 + e1;
#pragma unroll
      for (int msk = 1; msk <= 8; msk <<= 1) rs += __shfl_xor(rs, msk, 32);
      const float alpha = __expf(mi[i] - mnew);
      li[i] = li[i] * alpha + rs;
      mi[i] = mnew;
      acc[0][i] *= alpha; acc[1][i] *= alpha; acc[2][i] *= alpha; acc[3][i] *= alpha;
      // C layout -> LDS [row][key-in-block], bf16
      Pl[wave][(i + 8 * hi) * 32 + ln]      = (bf16_t)e0;
      Pl[wave][(i + 8 * hi) * 32 + 16 + ln] = (bf16_t)e1;
    }
    // DS ops are in-order per wave: the A-frag loads below see the stores above.
    const v16bf pf = load_frag(&Pl[wave][ln * 32], hi);

    // ---- O += P @ V   (V transposed: contiguous along keys)
#pragma unroll
    for (int j = 0; j < 4; ++j) {
      const bf16_t* vp = vbase + (size_t)(16 * j + ln) * S + k0;
      acc[j] = wmma_bf16(pf, load_frag(vp, hi), acc[j]);
    }
  }

#pragma unroll
  for (int j = 0; j < 4; ++j) {
#pragma unroll
    for (int i = 0; i < 8; ++i) {
      const int m = q0 + i + 8 * hi;
      const int d = 16 * j + ln;
      ob[((size_t)(b * S + m) * NHEAD + h) * HEAD_DIM + d] = (bf16_t)(acc[j][i] / li[i]);
    }
  }
}

// ---------------------------------------------------------------------------
// SwiGLU gate: ff[m,n] = u * v * sigmoid(v), bf16 out
// ---------------------------------------------------------------------------
__global__ void k_swiglu_bf16(const float* __restrict__ uv, bf16_t* __restrict__ ff, int n) {
  int i = blockIdx.x * blockDim.x + threadIdx.x;
  if (i >= n) return;
  const int m = i / D_FF;
  const int c = i - m * D_FF;
  const float u = uv[(size_t)m * (2 * D_FF) + c];
  const float g = uv[(size_t)m * (2 * D_FF) + D_FF + c];
  ff[i] = (bf16_t)(u * g / (1.0f + __expf(-g)));
}

// ---------------------------------------------------------------------------
// Launch
// ---------------------------------------------------------------------------
extern "C" void kernel_launch(void* const* d_in, const int* in_sizes, int n_in,
                              void* d_out, int out_size, void* d_ws, size_t ws_size,
                              hipStream_t stream) {
  const float* x     = (const float*)d_in[0];
  const float* fcos  = (const float*)d_in[1];
  const float* fsin  = (const float*)d_in[2];
  const float* wq    = (const float*)d_in[3];
  const float* bq    = (const float*)d_in[4];
  const float* wk    = (const float*)d_in[5];
  const float* bk    = (const float*)d_in[6];
  const float* wv    = (const float*)d_in[7];
  const float* bv    = (const float*)d_in[8];
  const float* wo    = (const float*)d_in[9];
  const float* bo    = (const float*)d_in[10];
  const float* n1w   = (const float*)d_in[11];
  const float* n2w   = (const float*)d_in[12];
  const float* w_in  = (const float*)d_in[13];
  const float* b_in  = (const float*)d_in[14];
  const float* w_out = (const float*)d_in[15];
  const float* b_out = (const float*)d_in[16];
  float* out = (float*)d_out;

  char* ws = (char*)d_ws;
  const size_t MB = 1ull << 20;
  bf16_t* wq_b   = (bf16_t*)(ws + 0);          //  2 MB
  bf16_t* wk_b   = (bf16_t*)(ws + 2 * MB);     //  2 MB
  bf16_t* wv_b   = (bf16_t*)(ws + 4 * MB);     //  2 MB
  bf16_t* wo_b   = (bf16_t*)(ws + 6 * MB);     //  2 MB
  bf16_t* win_b  = (bf16_t*)(ws + 8 * MB);     // 16 MB
  bf16_t* wout_b = (bf16_t*)(ws + 24 * MB);    //  8 MB
  bf16_t* h_b    = (bf16_t*)(ws + 32 * MB);    //  8 MB
  float*  x1     = (float*)(ws + 40 * MB);     // 16 MB (live to end)
  bf16_t* h2_b   = (bf16_t*)(ws + 56 * MB);    //  8 MB
  bf16_t* ff_b   = (bf16_t*)(ws + 64 * MB);    // 32 MB
  float*  q_f    = (float*)(ws + 96 * MB);     // 16 MB  (dead after rope)
  float*  k_f    = (float*)(ws + 112 * MB);    // 16 MB
  float*  v_f    = (float*)(ws + 128 * MB);    // 16 MB
  bf16_t* q_b    = (bf16_t*)(ws + 144 * MB);   //  8 MB  (dead after attn)
  bf16_t* k_b    = (bf16_t*)(ws + 152 * MB);   //  8 MB
  bf16_t* vt_b   = (bf16_t*)(ws + 160 * MB);   //  8 MB
  bf16_t* o_b    = (bf16_t*)(ws + 168 * MB);   //  8 MB  (dead after o-proj)
  float*  uv     = (float*)(ws + 96 * MB);     // 128 MB, overlaps dead qkv scratch

  const int T = 256;
  auto cdiv = [](int a, int b) { return (a + b - 1) / b; };

  // 1) weights -> bf16
  k_convert_bf16<<<cdiv(D_MODEL * D_MODEL, T), T, 0, stream>>>(wq, wq_b, D_MODEL * D_MODEL);
  k_convert_bf16<<<cdiv(D_MODEL * D_MODEL, T), T, 0, stream>>>(wk, wk_b, D_MODEL * D_MODEL);
  k_convert_bf16<<<cdiv(D_MODEL * D_MODEL, T), T, 0, stream>>>(wv, wv_b, D_MODEL * D_MODEL);
  k_convert_bf16<<<cdiv(D_MODEL * D_MODEL, T), T, 0, stream>>>(wo, wo_b, D_MODEL * D_MODEL);
  k_convert_bf16<<<cdiv(2 * D_FF * D_MODEL, T), T, 0, stream>>>(w_in, win_b, 2 * D_FF * D_MODEL);
  k_convert_bf16<<<cdiv(D_MODEL * D_FF, T), T, 0, stream>>>(w_out, wout_b, D_MODEL * D_FF);

  // 2) h = rmsnorm(x) -> bf16
  k_rmsnorm_bf16<<<Mrows, T, 0, stream>>>(x, n1w, h_b, D_MODEL);

  // 3) QKV projections (WMMA); each block: 256 rows x 64 cols
  dim3 g1(D_MODEL / 64, Mrows / 256);
  k_gemm_bf16_wmma<<<g1, T, 0, stream>>>(h_b, wq_b, bq, nullptr, q_f, Mrows, D_MODEL, D_MODEL);
  k_gemm_bf16_wmma<<<g1, T, 0, stream>>>(h_b, wk_b, bk, nullptr, k_f, Mrows, D_MODEL, D_MODEL);
  k_gemm_bf16_wmma<<<g1, T, 0, stream>>>(h_b, wv_b, bv, nullptr, v_f, Mrows, D_MODEL, D_MODEL);

  // 4) RoPE(q,k) -> bf16; V -> bf16 transposed [B,H,Dh,S]
  const int npairs = Bsz * S * NHEAD * (HEAD_DIM / 2);
  k_rope_bf16<<<cdiv(npairs, T), T, 0, stream>>>(q_f, fcos, fsin, q_b, npairs);
  k_rope_bf16<<<cdiv(npairs, T), T, 0, stream>>>(k_f, fcos, fsin, k_b, npairs);
  k_v_transpose_bf16<<<cdiv(Mrows * D_MODEL, T), T, 0, stream>>>(v_f, vt_b, Mrows * D_MODEL);

  // 5) causal flash attention (WMMA), one wave per 16-query tile
  const int nwaves = Bsz * NHEAD * (S / 16);   // 4096
  k_attn_wmma<<<nwaves / 8, T, 0, stream>>>(q_b, k_b, vt_b, o_b);

  // 6) x1 = x + attn @ wo^T + bo
  k_gemm_bf16_wmma<<<g1, T, 0, stream>>>(o_b, wo_b, bo, x, x1, Mrows, D_MODEL, D_MODEL);

  // 7) h2 = rmsnorm(x1) -> bf16
  k_rmsnorm_bf16<<<Mrows, T, 0, stream>>>(x1, n2w, h2_b, D_MODEL);

  // 8) uv = h2 @ w_in^T + b_in
  dim3 g2(2 * D_FF / 64, Mrows / 256);
  k_gemm_bf16_wmma<<<g2, T, 0, stream>>>(h2_b, win_b, b_in, nullptr, uv, Mrows, 2 * D_FF, D_MODEL);

  // 9) ff = u * silu(v) -> bf16
  k_swiglu_bf16<<<cdiv(Mrows * D_FF, T), T, 0, stream>>>(uv, ff_b, Mrows * D_FF);

  // 10) out = x1 + ff @ w_out^T + b_out
  dim3 g3(D_MODEL / 64, Mrows / 256);
  k_gemm_bf16_wmma<<<g3, T, 0, stream>>>(ff_b, wout_b, b_out, x1, out, Mrows, D_MODEL, D_FF);

  (void)in_sizes; (void)n_in; (void)out_size; (void)ws_size;
}
